// AttentionBlock_37374805409869
// MI455X (gfx1250) — matrix operand order repaired and checked by reading, not compiled, not run on hardware
//
#include <hip/hip_runtime.h>
#include <hip/hip_bf16.h>
#include <math.h>

#define B_   2
#define C_   512
#define N_   4096
#define H_   8
#define D_   64
#define G_   32
#define CPG  16
#define OQKV 1536

typedef __bf16 bf16_t;
typedef __attribute__((ext_vector_type(16))) __bf16 v16bf;
typedef __attribute__((ext_vector_type(8)))  __bf16 v8bf;
typedef __attribute__((ext_vector_type(8)))  float  v8f;

__device__ __forceinline__ bf16_t to_bf16(float f) {
    unsigned u = __builtin_bit_cast(unsigned, f);
    unsigned r = u + 0x7FFFu + ((u >> 16) & 1u);           // round-to-nearest-even
    unsigned short h = (unsigned short)(r >> 16);
    return __builtin_bit_cast(bf16_t, h);
}

__device__ __forceinline__ v16bf combine8(v8bf lo, v8bf hi) {
    return __builtin_shufflevector(lo, hi, 0,1,2,3,4,5,6,7,8,9,10,11,12,13,14,15);
}

// A-matrix fragment (16x32 bf16): lane row = lane%16; K elements are one
// contiguous run of 16: {half*16 .. +15}
__device__ __forceinline__ v16bf load_a(const bf16_t* rowptr, int kbase, int half) {
    v8bf lo = *reinterpret_cast<const v8bf*>(rowptr + kbase + half * 16);
    v8bf hi = *reinterpret_cast<const v8bf*>(rowptr + kbase + half * 16 + 8);
    return combine8(lo, hi);
}

// CDNA5 async copy: global -> LDS, 16 bytes per lane, tracked by ASYNCcnt.
// lds_off is a byte offset from the workgroup LDS base (dynamic LDS => base 0).
__device__ __forceinline__ void async_copy_b128(unsigned lds_off, const void* gptr) {
    unsigned long long ga = (unsigned long long)gptr;
    asm volatile("global_load_async_to_lds_b128 %0, %1, off"
                 :: "v"(lds_off), "v"(ga) : "memory");
}
__device__ __forceinline__ void wait_async0() {
    asm volatile("s_wait_asynccnt 0x0" ::: "memory");
}

// ---------------- weight fp32 -> bf16 ----------------
__global__ void wconv_kernel(const float* __restrict__ qw, const float* __restrict__ pw,
                             bf16_t* __restrict__ qwb, bf16_t* __restrict__ pwb) {
    const int NQ = OQKV * C_;
    const int NP = C_ * C_;
    for (int i = blockIdx.x * blockDim.x + threadIdx.x; i < NQ + NP;
         i += gridDim.x * blockDim.x) {
        if (i < NQ) qwb[i] = to_bf16(qw[i]);
        else        pwb[i - NQ] = to_bf16(pw[i - NQ]);
    }
}

// ---------------- GroupNorm(32, C) -> hn[b][n][c] bf16 ----------------
__global__ void gn_kernel(const float* __restrict__ x, const float* __restrict__ nw,
                          const float* __restrict__ nb, bf16_t* __restrict__ hn) {
    const int b = blockIdx.x / G_;
    const int g = blockIdx.x % G_;
    const int c0 = g * CPG;
    const int tid = threadIdx.x;
    const float* xb = x + (size_t)b * C_ * N_;

    float s = 0.f, s2 = 0.f;
    for (int i = tid; i < CPG * N_; i += 256) {
        int c = c0 + (i >> 12);
        int n = i & (N_ - 1);
        float v = xb[(size_t)c * N_ + n];
        s += v; s2 += v * v;
    }
    __shared__ float rs[256], rs2[256];
    rs[tid] = s; rs2[tid] = s2;
    __syncthreads();
    for (int off = 128; off > 0; off >>= 1) {
        if (tid < off) { rs[tid] += rs[tid + off]; rs2[tid] += rs2[tid + off]; }
        __syncthreads();
    }
    const float inv  = 1.0f / (float)(CPG * N_);
    const float mean = rs[0] * inv;
    const float var  = rs2[0] * inv - mean * mean;
    const float rstd = rsqrtf(var + 1e-5f);

    for (int i = tid; i < CPG * N_; i += 256) {
        int c = c0 + (i >> 12);
        int n = i & (N_ - 1);
        float v = xb[(size_t)c * N_ + n];
        float y = (v - mean) * rstd * nw[c] + nb[c];
        hn[((size_t)b * N_ + n) * C_ + c] = to_bf16(y);
    }
}

// ---------------- QKV GEMM: one wave per 16(o) x 64(n) tile ----------------
// A fragment reused across 4 column sub-tiles -> 2.5 b128 loads per wmma.
__global__ void qkv_gemm_kernel(const bf16_t* __restrict__ wq, const bf16_t* __restrict__ hn,
                                const float* __restrict__ qb,
                                bf16_t* __restrict__ qbuf, bf16_t* __restrict__ kbuf,
                                bf16_t* __restrict__ vbuf) {
    const int wave = threadIdx.x >> 5;
    const int lane = threadIdx.x & 31;
    const int half = lane >> 4;
    const int l16  = lane & 15;
    int tile = blockIdx.x * 4 + wave;                 // 2 * 96 * 64 tiles
    const int b = tile / (96 * 64);
    tile -= b * 96 * 64;
    const int ot = tile / 64;
    const int ng = tile % 64;

    const bf16_t* arow = wq + (size_t)(ot * 16 + l16) * C_;
    const bf16_t* brow0 = hn + ((size_t)b * N_ + ng * 64 + l16) * C_;

    v8f acc0 = {}, acc1 = {}, acc2 = {}, acc3 = {};
#pragma unroll
    for (int kk = 0; kk < C_ / 32; ++kk) {
        const int kb = kk * 32;
        v16bf a  = load_a(arow, kb, half);
        v16bf b0 = *reinterpret_cast<const v16bf*>(brow0 + 0 * 16 * C_ + kb + half * 16);
        v16bf b1 = *reinterpret_cast<const v16bf*>(brow0 + 1 * 16 * C_ + kb + half * 16);
        v16bf b2 = *reinterpret_cast<const v16bf*>(brow0 + 2 * 16 * C_ + kb + half * 16);
        v16bf b3 = *reinterpret_cast<const v16bf*>(brow0 + 3 * 16 * C_ + kb + half * 16);
        acc0 = __builtin_amdgcn_wmma_f32_16x16x32_bf16(false, a, false, b0, (short)0, acc0, false, false);
        acc1 = __builtin_amdgcn_wmma_f32_16x16x32_bf16(false, a, false, b1, (short)0, acc1, false, false);
        acc2 = __builtin_amdgcn_wmma_f32_16x16x32_bf16(false, a, false, b2, (short)0, acc2, false, false);
        acc3 = __builtin_amdgcn_wmma_f32_16x16x32_bf16(false, a, false, b3, (short)0, acc3, false, false);
    }
    v8f* accs[4] = { &acc0, &acc1, &acc2, &acc3 };
#pragma unroll
    for (int j = 0; j < 4; ++j) {
        const int n = ng * 64 + j * 16 + l16;
#pragma unroll
        for (int v = 0; v < 8; ++v) {                 // C/D row = v + half*8
            const int o = ot * 16 + v + half * 8;
            bf16_t bv = to_bf16((*accs[j])[v] + qb[o]);
            if (o < 512) {
                int h = o >> 6, d = o & 63;
                qbuf[(((size_t)b * H_ + h) * N_ + n) * D_ + d] = bv;
            } else if (o < 1024) {
                int o2 = o - 512, h = o2 >> 6, d = o2 & 63;
                kbuf[(((size_t)b * H_ + h) * N_ + n) * D_ + d] = bv;
            } else {
                int o2 = o - 1024, h = o2 >> 6, d = o2 & 63;
                vbuf[(((size_t)b * H_ + h) * D_ + d) * N_ + n] = bv;   // V transposed
            }
        }
    }
}

// ---------------- flash attention ----------------
// Block = 4 waves, same (b,h), adjacent 16-row tiles (64 rows / block).
// Per 64-column step: K panel (64x64) + V panel (64x64) staged once per block
// into LDS with global_load_async_to_lds_b128 (ASYNCcnt), then consumed by all
// 4 waves via ds_load_b128 B-fragments. Dynamic LDS => asm LDS offsets = byte idx.
#define KT_E 0          // Kt[mlocal][d]   : 64*64 bf16 (8 KB)
#define VT_E 4096       // Vt[d][mlocal]   : 64*64 bf16 (8 KB)
#define PT_E 8192       // Pt[wave][16][64]: 4*1024 bf16 (8 KB)

__global__ void attn_kernel(const bf16_t* __restrict__ qbuf, const bf16_t* __restrict__ kbuf,
                            const bf16_t* __restrict__ vbuf, bf16_t* __restrict__ ao) {
    extern __shared__ bf16_t smem[];
    const int wave = threadIdx.x >> 5;
    const int lane = threadIdx.x & 31;
    const int half = lane >> 4;
    const int l16  = lane & 15;
    const int b  = blockIdx.x / (H_ * 64);            // 2*8*64 blocks
    const int rm = blockIdx.x % (H_ * 64);
    const int h  = rm / 64;
    const int ng = rm % 64;
    const int nt = ng * 4 + wave;                     // this wave's 16-row tile
    const size_t bh = (size_t)b * H_ + h;

    const bf16_t* qrow = qbuf + (bh * N_ + nt * 16 + l16) * D_;
    v16bf qa0 = load_a(qrow, 0, half);
    v16bf qa1 = load_a(qrow, 32, half);

    float mrow[8], lrow[8], corr[8];
    v8f oacc[4];
#pragma unroll
    for (int v = 0; v < 8; ++v) { mrow[v] = -1e30f; lrow[v] = 0.f; }
#pragma unroll
    for (int dt = 0; dt < 4; ++dt) oacc[dt] = (v8f){};

    const float scale = 0.125f;                       // 1/sqrt(64)
    const int vrow = wave * 16 + (lane >> 3);         // V staging: 16 d-rows per wave
    const int vseg = (lane & 7) * 16;                 // byte segment within 128B row

    for (int mb = 0; mb < N_; mb += 64) {
        // ---- stage K (contiguous 8KB) and V (64 rows x 128B) into LDS ----
        const char* kg = (const char*)(kbuf + (bh * N_ + mb) * D_);
#pragma unroll
        for (int i = 0; i < 4; ++i) {
            unsigned off = (unsigned)(wave * 2048 + i * 512 + lane * 16);
            async_copy_b128(KT_E * 2 + off, kg + off);
        }
#pragma unroll
        for (int i = 0; i < 4; ++i) {
            int d = vrow + i * 4;
            async_copy_b128(VT_E * 2 + (unsigned)(d * 128 + vseg),
                            (const char*)(vbuf + (bh * D_ + d) * N_ + mb) + vseg);
        }
        wait_async0();
        __syncthreads();

        // ---- S = Q K^T for 4 column sub-tiles (K=64 over d) ----
        v8f s[4];
#pragma unroll
        for (int st = 0; st < 4; ++st) {
            const bf16_t* krow = smem + KT_E + (size_t)(st * 16 + l16) * 64;
            v16bf bk0 = *reinterpret_cast<const v16bf*>(krow + half * 16);
            v16bf bk1 = *reinterpret_cast<const v16bf*>(krow + 32 + half * 16);
            v8f t = {};
            t = __builtin_amdgcn_wmma_f32_16x16x32_bf16(false, qa0, false, bk0, (short)0, t, false, false);
            t = __builtin_amdgcn_wmma_f32_16x16x32_bf16(false, qa1, false, bk1, (short)0, t, false, false);
            s[st] = t;
        }

        // ---- online softmax across 64 columns ----
        bf16_t* prow = smem + PT_E + (size_t)wave * 1024;
#pragma unroll
        for (int v = 0; v < 8; ++v) {
            float a0 = s[0][v] * scale, a1 = s[1][v] * scale;
            float a2 = s[2][v] * scale, a3 = s[3][v] * scale;
            float rm2 = fmaxf(fmaxf(a0, a1), fmaxf(a2, a3));
#pragma unroll
            for (int msk = 8; msk > 0; msk >>= 1)
                rm2 = fmaxf(rm2, __shfl_xor(rm2, msk, 16));
            float mnew = fmaxf(mrow[v], rm2);
            corr[v] = __expf(mrow[v] - mnew);
            float p0 = __expf(a0 - mnew), p1 = __expf(a1 - mnew);
            float p2 = __expf(a2 - mnew), p3 = __expf(a3 - mnew);
            float ps = (p0 + p1) + (p2 + p3);
#pragma unroll
            for (int msk = 8; msk > 0; msk >>= 1)
                ps += __shfl_xor(ps, msk, 16);
            lrow[v] = lrow[v] * corr[v] + ps;
            mrow[v] = mnew;
            bf16_t* pr = prow + (size_t)(v + half * 8) * 64 + l16;
            pr[0]  = to_bf16(p0);
            pr[16] = to_bf16(p1);
            pr[32] = to_bf16(p2);
            pr[48] = to_bf16(p3);
        }
        __syncthreads();

        // ---- O += P V  (K=64 over m, two A-fragments from LDS) ----
        const bf16_t* parow = prow + (size_t)l16 * 64;
        v16bf pa0 = load_a(parow, 0, half);
        v16bf pa1 = load_a(parow, 32, half);
#pragma unroll
        for (int dt = 0; dt < 4; ++dt) {
            const bf16_t* vr = smem + VT_E + (size_t)(dt * 16 + l16) * 64;
            v16bf vb0 = *reinterpret_cast<const v16bf*>(vr + half * 16);
            v16bf vb1 = *reinterpret_cast<const v16bf*>(vr + 32 + half * 16);
            v8f c = oacc[dt];
#pragma unroll
            for (int v = 0; v < 8; ++v) c[v] *= corr[v];
            c = __builtin_amdgcn_wmma_f32_16x16x32_bf16(false, pa0, false, vb0, (short)0, c, false, false);
            c = __builtin_amdgcn_wmma_f32_16x16x32_bf16(false, pa1, false, vb1, (short)0, c, false, false);
            oacc[dt] = c;
        }
        __syncthreads();                               // panel consumed; safe to restage
    }
    // finalize: ao[b][n][h*64+d] bf16 (B-operand layout for proj GEMM)
#pragma unroll
    for (int v = 0; v < 8; ++v) {
        const float invl = 1.0f / lrow[v];
        const int n = nt * 16 + v + half * 8;
#pragma unroll
        for (int dt = 0; dt < 4; ++dt) {
            int d = dt * 16 + l16;
            ao[((size_t)b * N_ + n) * C_ + h * D_ + d] = to_bf16(oacc[dt][v] * invl);
        }
    }
}

// ---------------- proj GEMM + bias + residual: 16(o) x 64(n) per wave -------
__global__ void proj_kernel(const bf16_t* __restrict__ wp, const bf16_t* __restrict__ ao,
                            const float* __restrict__ pb, const float* __restrict__ x,
                            float* __restrict__ out) {
    const int wave = threadIdx.x >> 5;
    const int lane = threadIdx.x & 31;
    const int half = lane >> 4;
    const int l16  = lane & 15;
    int tile = blockIdx.x * 4 + wave;                 // 2 * 32 * 64 tiles
    const int b = tile / (32 * 64);
    tile -= b * 32 * 64;
    const int ot = tile / 64;
    const int ng = tile % 64;

    const bf16_t* arow  = wp + (size_t)(ot * 16 + l16) * C_;
    const bf16_t* brow0 = ao + ((size_t)b * N_ + ng * 64 + l16) * C_;

    v8f acc0 = {}, acc1 = {}, acc2 = {}, acc3 = {};
#pragma unroll
    for (int kk = 0; kk < C_ / 32; ++kk) {
        const int kb = kk * 32;
        v16bf a  = load_a(arow, kb, half);
        v16bf b0 = *reinterpret_cast<const v16bf*>(brow0 + 0 * 16 * C_ + kb + half * 16);
        v16bf b1 = *reinterpret_cast<const v16bf*>(brow0 + 1 * 16 * C_ + kb + half * 16);
        v16bf b2 = *reinterpret_cast<const v16bf*>(brow0 + 2 * 16 * C_ + kb + half * 16);
        v16bf b3 = *reinterpret_cast<const v16bf*>(brow0 + 3 * 16 * C_ + kb + half * 16);
        acc0 = __builtin_amdgcn_wmma_f32_16x16x32_bf16(false, a, false, b0, (short)0, acc0, false, false);
        acc1 = __builtin_amdgcn_wmma_f32_16x16x32_bf16(false, a, false, b1, (short)0, acc1, false, false);
        acc2 = __builtin_amdgcn_wmma_f32_16x16x32_bf16(false, a, false, b2, (short)0, acc2, false, false);
        acc3 = __builtin_amdgcn_wmma_f32_16x16x32_bf16(false, a, false, b3, (short)0, acc3, false, false);
    }
    v8f* accs[4] = { &acc0, &acc1, &acc2, &acc3 };
#pragma unroll
    for (int j = 0; j < 4; ++j) {
        const int n = ng * 64 + j * 16 + l16;
#pragma unroll
        for (int v = 0; v < 8; ++v) {
            const int o = ot * 16 + v + half * 8;
            size_t idx = ((size_t)b * C_ + o) * N_ + n;
            out[idx] = (*accs[j])[v] + pb[o] + x[idx];
        }
    }
}

extern "C" void kernel_launch(void* const* d_in, const int* in_sizes, int n_in,
                              void* d_out, int out_size, void* d_ws, size_t ws_size,
                              hipStream_t stream) {
    (void)in_sizes; (void)n_in; (void)out_size; (void)ws_size;
    const float* x  = (const float*)d_in[0];
    const float* nw = (const float*)d_in[1];
    const float* nb = (const float*)d_in[2];
    const float* qw = (const float*)d_in[3];
    const float* qb = (const float*)d_in[4];
    const float* pw = (const float*)d_in[5];
    const float* pb = (const float*)d_in[6];
    float* out = (float*)d_out;

    char* ws = (char*)d_ws;
    bf16_t* hn   = (bf16_t*)(ws + 0);                          //  8 MB  [b][n][c]
    bf16_t* qbuf = (bf16_t*)(ws + (size_t)( 8u << 20));        //  8 MB  [b][h][n][d]
    bf16_t* kbuf = (bf16_t*)(ws + (size_t)(16u << 20));        //  8 MB  [b][h][n][d]
    bf16_t* vbuf = (bf16_t*)(ws + (size_t)(24u << 20));        //  8 MB  [b][h][d][n]
    bf16_t* qwb  = (bf16_t*)(ws + (size_t)(32u << 20));        //  1.5 MB
    bf16_t* pwb  = (bf16_t*)(ws + (size_t)(32u << 20) + (size_t)OQKV * C_ * 2);
    bf16_t* ao   = hn;                                         // reuse hn after QKV

    wconv_kernel<<<2048, 256, 0, stream>>>(qw, pw, qwb, pwb);
    gn_kernel<<<B_ * G_, 256, 0, stream>>>(x, nw, nb, hn);
    qkv_gemm_kernel<<<(B_ * 96 * 64) / 4, 128, 0, stream>>>(qwb, hn, qb, qbuf, kbuf, vbuf);
    attn_kernel<<<B_ * H_ * 64, 128, 24576, stream>>>(qbuf, kbuf, vbuf, ao);
    proj_kernel<<<(B_ * 32 * 64) / 4, 128, 0, stream>>>(pwb, ao, pb, x, out);
}
